// AFTransformerLayer_69784628626214
// MI455X (gfx1250) — matrix operand order repaired and checked by reading, not compiled, not run on hardware
//
#include <hip/hip_runtime.h>
#include <hip/hip_bf16.h>

// ---------- types ----------
typedef __attribute__((ext_vector_type(16))) __bf16 bf16x16;
typedef __attribute__((ext_vector_type(8)))  __bf16 bf16x8;
typedef __attribute__((ext_vector_type(8)))  float  f32x8;

static __device__ __forceinline__ __bf16 f2bf(float f) {
  // round-to-nearest-even f32 -> bf16
  unsigned u = __builtin_bit_cast(unsigned, f);
  unsigned r = (u + 0x7FFFu + ((u >> 16) & 1u)) >> 16;
  unsigned short s = (unsigned short)r;
  return __builtin_bit_cast(__bf16, s);
}

// ---------- gfx1250 async global->LDS copies (ASYNCcnt path), with sync fallback ----------
#if defined(__has_builtin)
#  if __has_builtin(__builtin_amdgcn_global_load_async_to_lds_b128)
#    define USE_ASYNC 1
#  endif
#endif
#ifndef USE_ASYNC
#  define USE_ASYNC 0
#endif

#if USE_ASYNC
typedef int v4i __attribute__((__vector_size__(16)));
typedef __attribute__((address_space(1))) void gvoid;
typedef __attribute__((address_space(3))) void lvoid;
typedef __attribute__((address_space(1))) v4i gv4i;
typedef __attribute__((address_space(3))) v4i lv4i;
static __device__ __forceinline__ void async_copy16(const __bf16* g, __bf16* l) {
  __builtin_amdgcn_global_load_async_to_lds_b128((gv4i*)(gvoid*)g, (lv4i*)(lvoid*)l, 0, 0);
}
static __device__ __forceinline__ void wait_async0() {
#  if __has_builtin(__builtin_amdgcn_s_wait_asynccnt)
  __builtin_amdgcn_s_wait_asynccnt(0);
#  else
  asm volatile("s_wait_asynccnt 0x0" ::: "memory");
#  endif
}
#else
static __device__ __forceinline__ void async_copy16(const __bf16* g, __bf16* l) {
  *(bf16x8*)l = *(const bf16x8*)g;   // global_load_b128 + ds_store_b128
}
static __device__ __forceinline__ void wait_async0() {}
#endif

// ---------- generic bf16 WMMA GEMM: C[M,N] = A[M,K] @ B^T[N,K]^T (+epilogue) ----------
// A   : row-major [M,K] bf16 (lda = row stride)
// BmT : row-major [N,K] bf16 (B pre-transposed; ldbt = row stride)
// Double-buffered LDS, async global->LDS staging, 8 waves, 128x128x32 block tile,
// each wave: 2x4 grid of v_wmma_f32_16x16x32_bf16.
#define BM 128
#define BN 128
#define BKK 32

enum { MODE_F32 = 0, MODE_SIG = 1, MODE_BF16 = 2, MODE_RELU_BF16 = 3, MODE_RES = 4 };

__global__ __launch_bounds__(256)
void gemm_bf16_wmma(const __bf16* __restrict__ A, const __bf16* __restrict__ BmT,
                    const float* __restrict__ bias, const float* __restrict__ res,
                    float* __restrict__ Cf, __bf16* __restrict__ Cb,
                    int K, int lda, int ldbt, int ldc,
                    long long sA, long long sB, long long sC, int mode)
{
  __shared__ __align__(16) __bf16 As[2][BM * BKK];   // row-major M x K tile
  __shared__ __align__(16) __bf16 Bt[2][BN * BKK];   // row-major N x K tile (B^T)

  const int tid   = threadIdx.x;
  const int lane  = tid & 31;
  const int wid   = tid >> 5;
  const int waveM = wid & 3;       // 0..3 -> 32-row strip
  const int waveN = wid >> 2;      // 0..1 -> 64-col strip
  const int lrow  = lane & 15;
  const int lhi   = lane >> 4;     // 0: lanes 0-15, 1: lanes 16-31
  const int kOffA = lhi ? 8  : 0;  // A fragment: K {0-7,16-23} vs {8-15,24-31}
  const int kOffB = lhi ? 8  : 0;  // B fragment: K {0-7,16-23} vs {8-15,24-31}

  const int nBase = blockIdx.x * BN;
  const int mBase = blockIdx.y * BM;
  const int z     = blockIdx.z;
  A   += (long long)z * sA;
  BmT += (long long)z * sB;
  const long long cOff = (long long)z * sC;

  // staging: each tile is 128 rows x 64B = 512 16B-chunks; 2 chunks/thread/tile
  const int sRow = tid >> 2;             // chunk row (this thread's first chunk)
  const int sC8  = (tid & 3) * 8;        // chunk col offset (elements)

  f32x8 acc[2][4] = {};

  const int nk = K / BKK;

  // ---- prologue: stage K-tile 0 into buffer 0 ----
  {
#pragma unroll
    for (int j = 0; j < 2; ++j) {
      const int row = sRow + j * 64;
      async_copy16(A   + (long long)(mBase + row) * lda  + sC8, &As[0][row * BKK + sC8]);
      async_copy16(BmT + (long long)(nBase + row) * ldbt + sC8, &Bt[0][row * BKK + sC8]);
    }
  }

  for (int ik = 0; ik < nk; ++ik) {
    const int buf = ik & 1;
    wait_async0();        // own async copies into `buf` complete
    __syncthreads();      // everyone's copies complete / prev reads done

    // ---- stage next K-tile into the other buffer (overlaps with WMMA below) ----
    if (ik + 1 < nk) {
      const int kk = (ik + 1) * BKK;
      const int nb = buf ^ 1;
#pragma unroll
      for (int j = 0; j < 2; ++j) {
        const int row = sRow + j * 64;
        async_copy16(A   + (long long)(mBase + row) * lda  + kk + sC8, &As[nb][row * BKK + sC8]);
        async_copy16(BmT + (long long)(nBase + row) * ldbt + kk + sC8, &Bt[nb][row * BKK + sC8]);
      }
    }

    // ---- fragments + 8 WMMAs from current buffer ----
    bf16x16 afrag[2], bfrag[4];
#pragma unroll
    for (int wm = 0; wm < 2; ++wm) {
      const __bf16* ap = &As[buf][(waveM * 32 + wm * 16 + lrow) * BKK + kOffA];
      bf16x8 lo = *(const bf16x8*)ap;
      bf16x8 hi = *(const bf16x8*)(ap + 16);
      afrag[wm] = __builtin_shufflevector(lo, hi, 0,1,2,3,4,5,6,7,8,9,10,11,12,13,14,15);
    }
#pragma unroll
    for (int wn = 0; wn < 4; ++wn) {
      const __bf16* bp = &Bt[buf][(waveN * 64 + wn * 16 + lrow) * BKK + kOffB];
      bf16x8 lo = *(const bf16x8*)bp;
      bf16x8 hi = *(const bf16x8*)(bp + 16);
      bfrag[wn] = __builtin_shufflevector(lo, hi, 0,1,2,3,4,5,6,7,8,9,10,11,12,13,14,15);
    }
#pragma unroll
    for (int wm = 0; wm < 2; ++wm)
#pragma unroll
      for (int wn = 0; wn < 4; ++wn)
        acc[wm][wn] = __builtin_amdgcn_wmma_f32_16x16x32_bf16(
            false, afrag[wm], false, bfrag[wn], (short)0, acc[wm][wn], false, false);
  }

  // epilogue: C/D layout -> VGPR r: lanes0-15 M=r, lanes16-31 M=r+8; N=lane%16
  const int mhalf = lhi * 8;
#pragma unroll
  for (int wm = 0; wm < 2; ++wm) {
#pragma unroll
    for (int wn = 0; wn < 4; ++wn) {
      const int col = nBase + waveN * 64 + wn * 16 + lrow;
      const float bv = bias ? bias[col] : 0.f;
#pragma unroll
      for (int r = 0; r < 8; ++r) {
        const int row = mBase + waveM * 32 + wm * 16 + mhalf + r;
        const long long idx = cOff + (long long)row * ldc + col;
        float v = acc[wm][wn][r] + bv;
        switch (mode) {
          case MODE_F32:       Cf[idx] = v; break;
          case MODE_SIG:       Cf[idx] = 1.f / (1.f + __expf(-v)); break;
          case MODE_BF16:      Cb[idx] = f2bf(v); break;
          case MODE_RELU_BF16: Cb[idx] = f2bf(v > 0.f ? v : 0.f); break;
          case MODE_RES:       Cf[idx] = v + res[idx]; break;
        }
      }
    }
  }
}

// ---------- LayerNorm (row-wise) -> bf16 ----------
__global__ __launch_bounds__(256)
void ln_bf16_kernel(const float* __restrict__ x, const float* __restrict__ g,
                    const float* __restrict__ b, __bf16* __restrict__ out, int D)
{
  const int row = blockIdx.x;
  const int tid = threadIdx.x;
  const float* xr = x + (long long)row * D;
  float s = 0.f, s2 = 0.f;
  for (int c = tid; c < D; c += 256) { float v = xr[c]; s += v; s2 += v * v; }
  __shared__ float rs[256], rs2[256];
  rs[tid] = s; rs2[tid] = s2; __syncthreads();
  for (int off = 128; off > 0; off >>= 1) {
    if (tid < off) { rs[tid] += rs[tid + off]; rs2[tid] += rs2[tid + off]; }
    __syncthreads();
  }
  const float mu   = rs[0] / D;
  const float var  = rs2[0] / D - mu * mu;
  const float rstd = rsqrtf(var + 1e-5f);
  __bf16* orow = out + (long long)row * D;
  for (int c = tid; c < D; c += 256)
    orow[c] = f2bf((xr[c] - mu) * rstd * g[c] + b[c]);
}

// ---------- tiled f32 -> bf16 transpose: W[K,N] -> WT[N,K] ----------
__global__ __launch_bounds__(256)
void cvt_transpose_kernel(const float* __restrict__ W, __bf16* __restrict__ WT, int K, int N)
{
  __shared__ float tile[32][33];
  const int tx = threadIdx.x;      // 0..31
  const int ty = threadIdx.y;      // 0..7
  const int nBase = blockIdx.x * 32;
  const int kBase = blockIdx.y * 32;
#pragma unroll
  for (int j = 0; j < 4; ++j)
    tile[ty + 8 * j][tx] = W[(long long)(kBase + ty + 8 * j) * N + nBase + tx];
  __syncthreads();
#pragma unroll
  for (int j = 0; j < 4; ++j)
    WT[(long long)(nBase + ty + 8 * j) * K + kBase + tx] = f2bf(tile[tx][ty + 8 * j]);
}

// ---------- build transposed KV: kvT[b][c][t]=exp(k)*v, kvT[b][D+c][t]=exp(k) ----------
__global__ __launch_bounds__(256)
void build_kvT_kernel(const float* __restrict__ kf, const float* __restrict__ vf,
                      __bf16* __restrict__ kvT, int D, int T)
{
  __shared__ float kt[32][33], vt[32][33];
  const int tx = threadIdx.x, ty = threadIdx.y;
  const int cBase = blockIdx.x * 32;
  const int tBase = blockIdx.y * 32;
  const int b = blockIdx.z;
  const long long srcRow0 = (long long)b * T + tBase;
#pragma unroll
  for (int j = 0; j < 4; ++j) {
    const long long s = (srcRow0 + ty + 8 * j) * D + cBase + tx;
    kt[ty + 8 * j][tx] = kf[s];
    vt[ty + 8 * j][tx] = vf[s];
  }
  __syncthreads();
  __bf16* base = kvT + (long long)b * 2 * D * T;
#pragma unroll
  for (int j = 0; j < 4; ++j) {
    const int c = cBase + ty + 8 * j;
    const float ek = __expf(kt[tx][ty + 8 * j]);
    const float vv = vt[tx][ty + 8 * j];
    base[(long long)c * T + tBase + tx]       = f2bf(ek * vv);
    base[(long long)(D + c) * T + tBase + tx] = f2bf(ek);
  }
}

// ---------- elementwise helpers ----------
__global__ void exp_bf16_kernel(const float* __restrict__ in, __bf16* __restrict__ out, long long n) {
  // max-stabilizer cancels exactly in the num/den ratio; pos_bias ~ 0.1 scale
  long long i = blockIdx.x * (long long)blockDim.x + threadIdx.x;
  const long long stride = (long long)gridDim.x * blockDim.x;
  for (; i < n; i += stride) out[i] = f2bf(__expf(in[i]));
}

__global__ void attn_res_kernel(const float* __restrict__ qsig, const float* __restrict__ nd,
                                const float* __restrict__ x, float* __restrict__ x2,
                                int D, long long n) {
  long long i = blockIdx.x * (long long)blockDim.x + threadIdx.x;
  const long long stride = (long long)gridDim.x * blockDim.x;
  for (; i < n; i += stride) {
    const long long row = i / D;
    const int col = (int)(i - row * D);
    const float num = nd[row * 2 * D + col];
    const float den = nd[row * 2 * D + D + col];
    x2[i] = qsig[i] * (num / den) + x[i];
  }
}

// ---------- host orchestration ----------
static void launch_gemm(const __bf16* A, const __bf16* BmT, const float* bias, const float* res,
                        float* Cf, __bf16* Cb, int M, int N, int K,
                        int lda, int ldbt, int ldc,
                        long long sA, long long sB, long long sC, int zdim, int mode,
                        hipStream_t stream) {
  dim3 grid(N / BN, M / BM, zdim);
  gemm_bf16_wmma<<<grid, 256, 0, stream>>>(A, BmT, bias, res, Cf, Cb,
                                           K, lda, ldbt, ldc, sA, sB, sC, mode);
}

extern "C" void kernel_launch(void* const* d_in, const int* in_sizes, int n_in,
                              void* d_out, int out_size, void* d_ws, size_t ws_size,
                              hipStream_t stream) {
  (void)in_sizes; (void)n_in; (void)out_size; (void)ws_size;
  const float* x   = (const float*)d_in[0];
  const float* Wq  = (const float*)d_in[1];
  const float* bq  = (const float*)d_in[2];
  const float* Wk  = (const float*)d_in[3];
  const float* bk  = (const float*)d_in[4];
  const float* Wv  = (const float*)d_in[5];
  const float* bv  = (const float*)d_in[6];
  const float* pb  = (const float*)d_in[7];
  const float* g1  = (const float*)d_in[8];
  const float* bl1 = (const float*)d_in[9];
  const float* W1  = (const float*)d_in[10];
  const float* b1  = (const float*)d_in[11];
  const float* W2  = (const float*)d_in[12];
  const float* b2  = (const float*)d_in[13];
  const float* g2  = (const float*)d_in[14];
  const float* bl2 = (const float*)d_in[15];

  constexpr int Bt_ = 8, T = 2048, D = 1024;
  constexpr long long BT = (long long)Bt_ * T;      // 16384 rows
  const long long MB_ = 1024ll * 1024ll;

  char* ws = (char*)d_ws;
  __bf16* hb   = (__bf16*)(ws + 0 * MB_);     // 32MB  [BT,D] ln1 out (reused as ln2 out)
  __bf16* ewb  = (__bf16*)(ws + 32 * MB_);    // 8MB   [T,T]  exp(pos_bias)  (A of attn GEMM)
  __bf16* wqT  = (__bf16*)(ws + 40 * MB_);    // 2MB   Wq^T [D,D]
  __bf16* wkT  = (__bf16*)(ws + 42 * MB_);    // 2MB
  __bf16* wvT  = (__bf16*)(ws + 44 * MB_);    // 2MB
  __bf16* w1T  = (__bf16*)(ws + 46 * MB_);    // 8MB   W1^T [4D,D]
  __bf16* w2T  = (__bf16*)(ws + 54 * MB_);    // 8MB   W2^T [D,4D]
  __bf16* kvT  = (__bf16*)(ws + 62 * MB_);    // 64MB  [B][2D][T] = [ek*v ; ek] transposed
  float*  qsig = (float*)(ws + 126 * MB_);    // 64MB  sigmoid(q)
  float*  kf   = (float*)(ws + 190 * MB_);    // 64MB  k
  float*  vf   = (float*)(ws + 254 * MB_);    // 64MB  v
  float*  nd   = (float*)(ws + 190 * MB_);    // 128MB [BT,2D] (reuses kf/vf after build_kvT)
  float*  x2   = (float*)(ws + 318 * MB_);    // 64MB  attention residual output
  __bf16* h2b  = hb;                          // reuse after QKV GEMMs
  __bf16* mid  = (__bf16*)(ws + 62 * MB_);    // 128MB (reuses kvT/qsig after attention)

  dim3 tb(32, 8);

  // 1) weights f32 -> bf16, pre-transposed so all GEMM B-operands are [N,K]
  cvt_transpose_kernel<<<dim3(D / 32, D / 32), tb, 0, stream>>>(Wq, wqT, D, D);
  cvt_transpose_kernel<<<dim3(D / 32, D / 32), tb, 0, stream>>>(Wk, wkT, D, D);
  cvt_transpose_kernel<<<dim3(D / 32, D / 32), tb, 0, stream>>>(Wv, wvT, D, D);
  cvt_transpose_kernel<<<dim3(4 * D / 32, D / 32), tb, 0, stream>>>(W1, w1T, D, 4 * D);
  cvt_transpose_kernel<<<dim3(D / 32, 4 * D / 32), tb, 0, stream>>>(W2, w2T, 4 * D, D);

  // 2) ew = exp(pos_bias) -> bf16
  exp_bf16_kernel<<<4096, 256, 0, stream>>>(pb, ewb, (long long)T * T);

  // 3) LN1 -> bf16
  ln_bf16_kernel<<<(int)BT, 256, 0, stream>>>(x, g1, bl1, hb, D);

  // 4-6) QKV projections (fused sigmoid for q)
  launch_gemm(hb, wqT, bq, nullptr, qsig, nullptr, (int)BT, D, D, D, D, D, 0, 0, 0, 1, MODE_SIG, stream);
  launch_gemm(hb, wkT, bk, nullptr, kf,   nullptr, (int)BT, D, D, D, D, D, 0, 0, 0, 1, MODE_F32, stream);
  launch_gemm(hb, wvT, bv, nullptr, vf,   nullptr, (int)BT, D, D, D, D, D, 0, 0, 0, 1, MODE_F32, stream);

  // 7) kvT = transpose([exp(k)*v | exp(k)]) -> bf16 [B][2D][T]
  build_kvT_kernel<<<dim3(D / 32, T / 32, Bt_), tb, 0, stream>>>(kf, vf, kvT, D, T);

  // 8) batched attention GEMM: nd[b][T,2D] = ew[T,T] @ kvT[b]^T
  launch_gemm(ewb, kvT, nullptr, nullptr, nd, nullptr,
              T, 2 * D, T, /*lda=*/T, /*ldbt=*/T, /*ldc=*/2 * D,
              /*sA=*/0, /*sB=*/(long long)2 * D * T, /*sC=*/(long long)T * 2 * D,
              Bt_, MODE_F32, stream);

  // 9) x2 = sigmoid(q) * num/den + x
  attn_res_kernel<<<8192, 256, 0, stream>>>(qsig, nd, x, x2, D, BT * D);

  // 10) LN2 -> bf16
  ln_bf16_kernel<<<(int)BT, 256, 0, stream>>>(x2, g2, bl2, h2b, D);

  // 11) mid = relu(h2 @ W1 + b1) -> bf16
  launch_gemm(h2b, w1T, b1, nullptr, nullptr, mid, (int)BT, 4 * D, D,
              D, D, 4 * D, 0, 0, 0, 1, MODE_RELU_BF16, stream);

  // 12) out = mid @ W2 + b2 + x2
  launch_gemm(mid, w2T, b2, x2, (float*)d_out, nullptr, (int)BT, D, 4 * D,
              4 * D, 4 * D, D, 0, 0, 0, 1, MODE_RES, stream);
}
